// UnsupervisedGAT_27771258536066
// MI455X (gfx1250) — matrix operand order repaired and checked by reading, not compiled, not run on hardware
//
#include <hip/hip_runtime.h>

#define NODES  100000
#define EDGES  1600000
#define FDIM   128      // HEADS * DHEAD
#define HEADS  8
#define DHEAD  16
#define CDIV(a,b) (((a)+(b)-1)/(b))

typedef __attribute__((ext_vector_type(16))) _Float16 v16h;
typedef __attribute__((ext_vector_type(8)))  _Float16 v8h;
typedef __attribute__((ext_vector_type(8)))  float    v8f;

// ---------------------------------------------------------------- utilities
__global__ __launch_bounds__(256) void fill_u32(unsigned* __restrict__ p, unsigned v, int n) {
  int i = blockIdx.x * 256 + threadIdx.x;
  if (i < n) p[i] = v;
}

// leaky_relu(slope) + f32->f16 convert (slope=1.0 => plain convert)
__global__ __launch_bounds__(256) void cvt_act_f16(const float* __restrict__ in,
                                                   _Float16* __restrict__ out,
                                                   float slope, int n) {
  int i = blockIdx.x * 256 + threadIdx.x;
  if (i >= n) return;
  float v = in[i];
  v = v >= 0.f ? v : slope * v;
  out[i] = (_Float16)v;
}

// Pack W[K][N] (f32, row-major, 128x128) into Wt[n][k] (f16).
// A lane of the GEMM then reads 16 contiguous halfs (32B) per B fragment.
__global__ __launch_bounds__(256) void pack_wt(const float* __restrict__ W,
                                               _Float16* __restrict__ Wt) {
  int i = blockIdx.x * 256 + threadIdx.x;   // 0 .. 128*128-1
  int k = i >> 7, n = i & 127;
  Wt[n * FDIM + k] = (_Float16)W[k * FDIM + n];
}

// ---------------------------------------------------------------- WMMA GEMM
// C[M][128] = A[M][128] (f16) * W[128][128], W given as Wt[n][k] f16.
// One wave per 16-row M-tile; wave computes all 8 N-tiles (16x128 stripe).
// Fragment layouts in contiguous K order, not the ISA table order (16-bit A 16x32; B K split across
// lane halves, 16 consecutive K per lane group).
__global__ __launch_bounds__(256) void gemm_wmma_f16(const _Float16* __restrict__ A,
                                                     const _Float16* __restrict__ Bt,
                                                     float* __restrict__ C,
                                                     int mtiles) {
  const int wave   = threadIdx.x >> 5;
  const int lane   = threadIdx.x & 31;
  const int mtile  = blockIdx.x * 8 + wave;
  if (mtile >= mtiles) return;              // uniform per wave: EXEC stays all-1s
  const int half16 = lane >> 4;             // lane group 0/1
  const int l16    = lane & 15;

  const _Float16* arow = A + (size_t)(mtile * 16 + l16) * FDIM;

  v8f zero = {};
  v8f acc[8];
#pragma unroll
  for (int t = 0; t < 8; ++t) acc[t] = zero;

#pragma unroll
  for (int k0 = 0; k0 < FDIM; k0 += 32) {
    // A 16x32 f16 fragment: lanes 0-15 hold K 0..15, lanes 16-31 K 16..31 (matches B)
    v16h a  = *(const v16h*)(arow + k0 + half16 * 16);
#pragma unroll
    for (int t = 0; t < 8; ++t) {
      // B 32x16: lane holds column (t*16+l16), K = k0 + 16*half16 .. +15 (contiguous in Wt)
      v16h b = *(const v16h*)(Bt + (size_t)(t * 16 + l16) * FDIM + k0 + half16 * 16);
      acc[t] = __builtin_amdgcn_wmma_f32_16x16x32_f16(
                   false, a, false, b, (short)0, acc[t], false, false);
    }
  }

  // C/D layout: VGPR v, lanes0-15 -> M=v, lanes16-31 -> M=v+8, N=l16
#pragma unroll
  for (int t = 0; t < 8; ++t) {
    float* cp = C + (size_t)(mtile * 16 + half16 * 8) * FDIM + t * 16 + l16;
#pragma unroll
    for (int v = 0; v < 8; ++v) cp[(size_t)v * FDIM] = acc[t][v];
  }
}

// ---------------------------------------------------------------- attention
// el[n,h] = sum_d h[n,h,d]*al[h,d] ; er likewise
__global__ __launch_bounds__(256) void node_attn(const float* __restrict__ h,
                                                 const float* __restrict__ al,
                                                 const float* __restrict__ ar,
                                                 float* __restrict__ el,
                                                 float* __restrict__ er, int nh) {
  int i = blockIdx.x * 256 + threadIdx.x;
  if (i >= nh) return;
  int hd = i & (HEADS - 1);
  const float* hp = h + (size_t)(i >> 3) * FDIM + hd * DHEAD;
  float sl = 0.f, sr = 0.f;
#pragma unroll
  for (int d = 0; d < DHEAD; ++d) {
    float v = hp[d];
    sl += v * al[hd * DHEAD + d];
    sr += v * ar[hd * DHEAD + d];
  }
  el[i] = sl;
  er[i] = sr;
}

// score = leaky_relu(el[src]+er[dst], 0.2); segment-max via order-preserving u32 encode
__global__ __launch_bounds__(256) void edge_score_max(const int* __restrict__ src,
                                                      const int* __restrict__ dst,
                                                      const float* __restrict__ el,
                                                      const float* __restrict__ er,
                                                      float* __restrict__ esc,
                                                      unsigned* __restrict__ emax, int eh) {
  int i = blockIdx.x * 256 + threadIdx.x;
  if (i >= eh) return;
  int e = i >> 3, hd = i & 7;
  int s = src[e], d = dst[e];
  float v = el[s * HEADS + hd] + er[d * HEADS + hd];
  v = v >= 0.f ? v : 0.2f * v;
  esc[i] = v;
  unsigned b   = __float_as_uint(v);
  unsigned enc = ((int)b >= 0) ? (b | 0x80000000u) : ~b;   // monotone map, min enc = 0
  atomicMax(&emax[d * HEADS + hd], enc);
}

// ee = exp(score - emax[dst]); denom[dst] += ee   (ee overwrites score in place)
__global__ __launch_bounds__(256) void edge_exp_sum(const int* __restrict__ dst,
                                                    float* __restrict__ esc,
                                                    const unsigned* __restrict__ emax,
                                                    float* __restrict__ denom, int eh) {
  int i = blockIdx.x * 256 + threadIdx.x;
  if (i >= eh) return;
  int e = i >> 3, hd = i & 7;
  int d = dst[e];
  unsigned u = emax[d * HEADS + hd];
  unsigned b = (u & 0x80000000u) ? (u & 0x7FFFFFFFu) : ~u;  // decode
  float m = __uint_as_float(b);
  float w = expf(esc[i] - m);
  esc[i] = w;
  atomicAdd(&denom[d * HEADS + hd], w);
}

// out[dst,f] += h[src,f] * ee[e,head]/denom[dst,head]  — L2-resident scatter-add
__global__ __launch_bounds__(256) void edge_aggregate(const int* __restrict__ src,
                                                      const int* __restrict__ dst,
                                                      const float* __restrict__ h,
                                                      const float* __restrict__ ee,
                                                      const float* __restrict__ denom,
                                                      float* __restrict__ out, int total) {
  int i = blockIdx.x * 256 + threadIdx.x;
  if (i >= total) return;                  // total = EDGES*FDIM = 204.8M (fits int)
  int e = i >> 7, f = i & 127, hd = f >> 4;
  int s = src[e], d = dst[e];
  float alpha = ee[e * HEADS + hd] / denom[d * HEADS + hd];
  atomicAdd(&out[(size_t)d * FDIM + f], h[(size_t)s * FDIM + f] * alpha);
}

// ---------------------------------------------------------------- launcher
extern "C" void kernel_launch(void* const* d_in, const int* in_sizes, int n_in,
                              void* d_out, int out_size, void* d_ws, size_t ws_size,
                              hipStream_t stream) {
  (void)in_sizes; (void)n_in; (void)out_size; (void)ws_size;
  const float* n_feat = (const float*)d_in[0];
  const int*   src    = (const int*)  d_in[1];
  const int*   dst    = (const int*)  d_in[2];
  const float* W1     = (const float*)d_in[3];
  const float* al1    = (const float*)d_in[4];
  const float* ar1    = (const float*)d_in[5];
  const float* W2     = (const float*)d_in[6];
  const float* al2    = (const float*)d_in[7];
  const float* ar2    = (const float*)d_in[8];

  char* ws = (char*)d_ws;
  size_t off = 0;
  auto take = [&](size_t bytes) -> char* {
    char* p = ws + off;
    off += (bytes + 255) & ~(size_t)255;
    return p;
  };

  float*    h_buf = (float*)   take((size_t)NODES * FDIM  * 4);  // 51.2 MB
  float*    agg   = (float*)   take((size_t)NODES * FDIM  * 4);  // 51.2 MB
  _Float16* xh    = (_Float16*)take((size_t)NODES * FDIM  * 2);  // 25.6 MB
  float*    ee    = (float*)   take((size_t)EDGES * HEADS * 4);  // 51.2 MB
  float*    el    = (float*)   take((size_t)NODES * HEADS * 4);
  float*    er    = (float*)   take((size_t)NODES * HEADS * 4);
  unsigned* emax  = (unsigned*)take((size_t)NODES * HEADS * 4);
  float*    denom = (float*)   take((size_t)NODES * HEADS * 4);
  _Float16* wt1   = (_Float16*)take((size_t)FDIM * FDIM * 2);
  _Float16* wt2   = (_Float16*)take((size_t)FDIM * FDIM * 2);

  const int NF = NODES * FDIM;    // 12.8M
  const int NH = NODES * HEADS;   // 0.8M
  const int EH = EDGES * HEADS;   // 12.8M
  const int EF = EDGES * FDIM;    // 204.8M
  const int MT = NODES / 16;      // 6250 M-tiles (N divisible by 16)

  // weight pre-pack (f16, WMMA-B-friendly layout) + input convert
  pack_wt<<<CDIV(FDIM * FDIM, 256), 256, 0, stream>>>(W1, wt1);
  pack_wt<<<CDIV(FDIM * FDIM, 256), 256, 0, stream>>>(W2, wt2);
  cvt_act_f16<<<CDIV(NF, 256), 256, 0, stream>>>(n_feat, xh, 1.0f, NF);

  // ---------------- layer 1 ----------------
  gemm_wmma_f16<<<CDIV(MT, 8), 256, 0, stream>>>(xh, wt1, h_buf, MT);
  node_attn<<<CDIV(NH, 256), 256, 0, stream>>>(h_buf, al1, ar1, el, er, NH);
  fill_u32<<<CDIV(NH, 256), 256, 0, stream>>>(emax, 0u, NH);
  fill_u32<<<CDIV(NH, 256), 256, 0, stream>>>((unsigned*)denom, 0u, NH);
  fill_u32<<<CDIV(NF, 256), 256, 0, stream>>>((unsigned*)agg, 0u, NF);
  edge_score_max<<<CDIV(EH, 256), 256, 0, stream>>>(src, dst, el, er, ee, emax, EH);
  edge_exp_sum<<<CDIV(EH, 256), 256, 0, stream>>>(dst, ee, emax, denom, EH);
  edge_aggregate<<<CDIV(EF, 256), 256, 0, stream>>>(src, dst, h_buf, ee, denom, agg, EF);

  // ---------------- layer 2 ----------------
  cvt_act_f16<<<CDIV(NF, 256), 256, 0, stream>>>(agg, xh, 0.01f, NF);  // activation + cvt
  gemm_wmma_f16<<<CDIV(MT, 8), 256, 0, stream>>>(xh, wt2, h_buf, MT);
  node_attn<<<CDIV(NH, 256), 256, 0, stream>>>(h_buf, al2, ar2, el, er, NH);
  fill_u32<<<CDIV(NH, 256), 256, 0, stream>>>(emax, 0u, NH);
  fill_u32<<<CDIV(NH, 256), 256, 0, stream>>>((unsigned*)denom, 0u, NH);
  fill_u32<<<CDIV(NF, 256), 256, 0, stream>>>((unsigned*)d_out, 0u, NF);
  edge_score_max<<<CDIV(EH, 256), 256, 0, stream>>>(src, dst, el, er, ee, emax, EH);
  edge_exp_sum<<<CDIV(EH, 256), 256, 0, stream>>>(dst, ee, emax, denom, EH);
  edge_aggregate<<<CDIV(EF, 256), 256, 0, stream>>>(src, dst, h_buf, ee, denom,
                                                    (float*)d_out, EF);
}
